// LSTM_77111842832812
// MI455X (gfx1250) — matrix operand order repaired and checked by reading, not compiled, not run on hardware
//
#include <hip/hip_runtime.h>
#include <hip/hip_bf16.h>

// ---------------------------------------------------------------------------
// LSTM for MI455X (gfx1250, wave32, WMMA).
//   H=512, IN=512, T=1024, B=64, gates 4H=2048 (order f,i,o,g)
//
//   K0a: pack Wx/Wh gate weights -> bf16, bsum=bx+bh, zero h ping-pong + cnt.
//   K0b: convert x -> bf16 once (avoids 16x redundant per-tile conversion).
//   K1 : xp(65536x2048) = x @ Wx^T + bsum. bf16 WMMA, fp32 accum.
//        A-tiles staged to LDS with double-buffered GLOBAL_LOAD_ASYNC_TO_LDS
//        (ASYNCcnt pipeline) when the builtin exists; sync copy otherwise.
//   K2 : persistent recurrence, 32 WGs. WG g owns H columns [16g,16g+16):
//        Wh slice in LDS feeds WMMA B-fragments, A = h from global ping-pong
//        bf16 buffer, fp32 gate math + cell state, one atomic grid barrier
//        per step, global_prefetch of next step's xp slice.
// ---------------------------------------------------------------------------

typedef __attribute__((ext_vector_type(16))) __bf16 v16bf;
typedef __attribute__((ext_vector_type(8)))  __bf16 v8bf;
typedef __attribute__((ext_vector_type(8)))  float  v8f;
typedef __attribute__((ext_vector_type(4)))  int    v4i;

#define TT   1024
#define BB   64
#define HH   512
#define NIN  512
#define NG   2048           // 4*H
#define MTOT (TT*BB)        // 65536
#define NWG2 32             // workgroups in recurrence kernel

// ---- workspace layout (bytes) ----
#define WS_WX   0u                          // 2 MB  Wx bf16 (2048x512)
#define WS_WH   (2u<<20)                    // 2 MB  Wh bf16 (2048x512)
#define WS_BS   (4u<<20)                    // 8 KB  bsum fp32 (2048)
#define WS_H    ((4u<<20) + (64u<<10))      // 128 KB h ping-pong bf16 (2x64x512)
#define WS_CNT  ((4u<<20) + (256u<<10))     // barrier counter
#define WS_XB   (8u<<20)                    // 64 MB x bf16 (65536x512)
#define WS_XP   (80u<<20)                   // 512 MB xp fp32 (65536x2048)

#if defined(__has_builtin)
#if __has_builtin(__builtin_amdgcn_global_load_async_to_lds_b128)
#define USE_ASYNC_COPY 1
#endif
#endif

#ifdef USE_ASYNC_COPY
// clang prints these param types as "int __vector(4) __device__*" /
// "... __shared__*", i.e. AS(1)/AS(3) pointers to 16-byte vectors.
typedef __attribute__((address_space(1))) v4i gv4i_t;
typedef __attribute__((address_space(3))) v4i lv4i_t;

static __device__ __forceinline__ void wait_async(int pending) {
#if __has_builtin(__builtin_amdgcn_s_wait_asynccnt)
  if (pending) __builtin_amdgcn_s_wait_asynccnt(2);
  else         __builtin_amdgcn_s_wait_asynccnt(0);
#else
  if (pending) asm volatile("s_wait_asynccnt 0x2" ::: "memory");
  else         asm volatile("s_wait_asynccnt 0x0" ::: "memory");
#endif
}
#endif

static __device__ __forceinline__ v16bf load_afrag(const __bf16* rowp, int lane) {
  // A 16x32 bf16: lanes 0-15 -> K {0..15}; lanes 16-31 -> K {16..31}
  const int kb = (lane >> 4) * 16;
  v8bf lo = *(const v8bf*)(rowp + kb);
  v8bf hi = *(const v8bf*)(rowp + kb + 8);
  return __builtin_shufflevector(lo, hi, 0,1,2,3,4,5,6,7,8,9,10,11,12,13,14,15);
}

static __device__ __forceinline__ float fsig(float x) {
  return 1.0f / (1.0f + __expf(-x));
}

// ---------------------------------------------------------------------------
// K0a: weight packing + state init
// ---------------------------------------------------------------------------
__global__ void __launch_bounds__(256) lstm_prep(
    const float* Wxf, const float* Whf, const float* Wxi, const float* Whi,
    const float* Wxo, const float* Who, const float* Wxg, const float* Whg,
    const float* bxf, const float* bhf, const float* bxi, const float* bhi,
    const float* bxo, const float* bho, const float* bxg, const float* bhg,
    __bf16* WxB, __bf16* WhB, float* bsum, __bf16* hping, unsigned* cnt) {
  const unsigned i = blockIdx.x * 256u + threadIdx.x;   // < 2048*512
  const float* wx[4] = {Wxf, Wxi, Wxo, Wxg};
  const float* wh[4] = {Whf, Whi, Who, Whg};
  if (i < (unsigned)NG * NIN) {
    const int row = i >> 9, k = i & 511;
    const int gate = row >> 9, r = row & 511;
    WxB[i] = (__bf16)wx[gate][r * NIN + k];
    WhB[i] = (__bf16)wh[gate][r * HH + k];
  }
  if (i < NG) {
    const float* bx[4] = {bxf, bxi, bxo, bxg};
    const float* bh[4] = {bhf, bhi, bho, bhg};
    const int gate = i >> 9, r = i & 511;
    bsum[i] = bx[gate][r] + bh[gate][r];
  }
  if (i < 2u * BB * HH) hping[i] = (__bf16)0.0f;        // zero both h buffers
  if (i == 0) *cnt = 0u;
}

// ---------------------------------------------------------------------------
// K0b: x fp32 -> bf16 (one pass, 8 elems/thread)
// ---------------------------------------------------------------------------
__global__ void __launch_bounds__(256) lstm_conv_x(
    const float* __restrict__ x, __bf16* __restrict__ xb) {
  const size_t i = ((size_t)blockIdx.x * 256u + threadIdx.x) * 8u;
  const float4 a = *(const float4*)(x + i);
  const float4 b = *(const float4*)(x + i + 4);
  v8bf o;
  o[0] = (__bf16)a.x; o[1] = (__bf16)a.y; o[2] = (__bf16)a.z; o[3] = (__bf16)a.w;
  o[4] = (__bf16)b.x; o[5] = (__bf16)b.y; o[6] = (__bf16)b.z; o[7] = (__bf16)b.w;
  *(v8bf*)(xb + i) = o;
}

// ---------------------------------------------------------------------------
// K1: xp = x @ Wx^T + bsum     (M=65536, N=2048, K=512), 128x128 tiles
// ---------------------------------------------------------------------------
__global__ void __launch_bounds__(256) lstm_xp_gemm(
    const __bf16* __restrict__ xb, const __bf16* __restrict__ WxB,
    const float* __restrict__ bsum, float* __restrict__ xp) {
  __shared__ __bf16 As[2][128 * 32];                    // 2 x 8 KB x-tiles
  const int tid  = threadIdx.x;
  const int lane = tid & 31, wave = tid >> 5;           // 8 waves
  const int m0 = blockIdx.x * 128, n0 = blockIdx.y * 128;

  v8f acc[8];
#pragma unroll
  for (int nt = 0; nt < 8; ++nt) acc[nt] = (v8f){0,0,0,0,0,0,0,0};

  // stage one 128x32 bf16 tile: 512 x 16-byte chunks, 2 per thread
  auto issue_copy = [&](int ib, int buf) {
    const int kb = ib * 32;
#pragma unroll
    for (int u = 0; u < 2; ++u) {
      const int c = tid + u * 256;                      // 0..511
      const int row = c >> 2, q = (c & 3) * 8;
      const __bf16* src = xb + (size_t)(m0 + row) * NIN + kb + q;
      __bf16* dst = &As[buf][row * 32 + q];
#ifdef USE_ASYNC_COPY
      __builtin_amdgcn_global_load_async_to_lds_b128((gv4i_t*)src, (lv4i_t*)dst, 0, 0);
#else
      *(v8bf*)dst = *(const v8bf*)src;
#endif
    }
  };

  issue_copy(0, 0);
  for (int ib = 0; ib < 16; ++ib) {
    const int kb = ib * 32;
    const int cur = ib & 1;
    if (ib + 1 < 16) issue_copy(ib + 1, cur ^ 1);       // prefetch next tile
#ifdef USE_ASYNC_COPY
    wait_async(ib + 1 < 16);                            // current tile landed
#endif
    __syncthreads();

    // A fragment for this wave's 16 M-rows
    const __bf16* arow = &As[cur][(wave * 16 + (lane & 15)) * 32];
    v16bf a = load_afrag(arow, lane);

    // B fragments straight from row-major bf16 Wx (K-major contiguous/lane)
#pragma unroll
    for (int nt = 0; nt < 8; ++nt) {
      const __bf16* bp = WxB + (size_t)(n0 + nt * 16 + (lane & 15)) * NIN
                         + kb + (lane >> 4) * 16;
      v16bf b = *(const v16bf*)bp;
      acc[nt] = __builtin_amdgcn_wmma_f32_16x16x32_bf16(
          false, a, false, b, (short)0, acc[nt], false, false);
    }
    __syncthreads();                                    // tile consumed
  }

  // store C + bias: VGPR r -> M = r + (lane/16)*8 ; N = lane&15
#pragma unroll
  for (int nt = 0; nt < 8; ++nt) {
    const int col = n0 + nt * 16 + (lane & 15);
    const float bb = bsum[col];
#pragma unroll
    for (int r = 0; r < 8; ++r) {
      const int m = m0 + wave * 16 + (lane >> 4) * 8 + r;
      xp[(size_t)m * NG + col] = acc[nt][r] + bb;
    }
  }
}

// ---------------------------------------------------------------------------
// K2: persistent recurrence, 32 WGs. WG g owns H columns [16g, 16g+16).
// ---------------------------------------------------------------------------
__global__ void __launch_bounds__(256, 1) lstm_recurrence(
    const float* __restrict__ xp, const __bf16* __restrict__ WhB,
    __bf16* __restrict__ hping, float* __restrict__ out,
    unsigned* __restrict__ cnt) {
  extern __shared__ char smem[];
  __bf16* WhS = (__bf16*)smem;                    // 64 rows x 512  (64 KB)
  float*  zbuf = (float*)(smem + 64 * 512 * 2);   // [gate][b][n]   (16 KB)
  float*  cbuf = zbuf + 4 * BB * 16;              // [b][n]          (4 KB)

  const int g = blockIdx.x;                       // 0..31
  const int tid = threadIdx.x, lane = tid & 31, wave = tid >> 5;

  // preload this WG's Wh slice: rows rr=gate*16+n -> Wh row gate*512+16g+n
  for (int idx = tid; idx < 64 * 512 / 8; idx += 256) {
    const int rr = idx >> 6, kc = (idx & 63) * 8;
    const int j = (rr >> 4) * HH + g * 16 + (rr & 15);
    *(v8bf*)&WhS[rr * 512 + kc] = *(const v8bf*)&WhB[(size_t)j * HH + kc];
  }
  for (int idx = tid; idx < BB * 16; idx += 256) cbuf[idx] = 0.0f;
  __syncthreads();

  const int mt = wave >> 1;                       // M-tile (batch 16-block)
  const int g0 = (wave & 1) * 2;                  // gates {g0, g0+1}

  for (int t = 0; t < TT; ++t) {
    const __bf16* hin  = hping + (size_t)(t & 1) * (BB * HH);
    __bf16*       hout = hping + (size_t)((t + 1) & 1) * (BB * HH);

    // prefetch next step's xp slice (64B segments; 256 segs = 64 b x 4 gates)
    if (t + 1 < TT) {
      const int b = tid >> 2, gate = tid & 3;
      __builtin_prefetch(xp + ((size_t)(t + 1) * BB + b) * NG + gate * 512 + g * 16, 0, 1);
    }

    // ---- z-slice GEMM: (64 x 512) @ Wh_slice^T, bf16 WMMA, fp32 accum
    v8f acc0 = (v8f){0,0,0,0,0,0,0,0};
    v8f acc1 = (v8f){0,0,0,0,0,0,0,0};
#pragma unroll 4
    for (int kb = 0; kb < HH; kb += 32) {
      const __bf16* arow = hin + (size_t)(mt * 16 + (lane & 15)) * HH + kb;
      v16bf a = load_afrag(arow, lane);
      const __bf16* b0p = &WhS[((g0 + 0) * 16 + (lane & 15)) * 512 + kb + (lane >> 4) * 16];
      const __bf16* b1p = &WhS[((g0 + 1) * 16 + (lane & 15)) * 512 + kb + (lane >> 4) * 16];
      v16bf b0 = *(const v16bf*)b0p;
      v16bf b1 = *(const v16bf*)b1p;
      acc0 = __builtin_amdgcn_wmma_f32_16x16x32_bf16(false, a, false, b0,
                                                     (short)0, acc0, false, false);
      acc1 = __builtin_amdgcn_wmma_f32_16x16x32_bf16(false, a, false, b1,
                                                     (short)0, acc1, false, false);
    }
    {
      const int n = lane & 15;
#pragma unroll
      for (int r = 0; r < 8; ++r) {
        const int m = mt * 16 + (lane >> 4) * 8 + r;
        zbuf[((g0 + 0) * BB + m) * 16 + n] = acc0[r];
        zbuf[((g0 + 1) * BB + m) * 16 + n] = acc1[r];
      }
    }
    __syncthreads();

    // ---- gates + state update (fp32), 4 elements/thread
    for (int i = tid; i < BB * 16; i += 256) {
      const int b = i >> 4, n = i & 15;
      const float* xpr = xp + ((size_t)t * BB + b) * NG + g * 16 + n;
      const float zf = zbuf[(0 * BB + b) * 16 + n] + xpr[0];
      const float zi = zbuf[(1 * BB + b) * 16 + n] + xpr[512];
      const float zo = zbuf[(2 * BB + b) * 16 + n] + xpr[1024];
      const float zg = zbuf[(3 * BB + b) * 16 + n] + xpr[1536];
      const float f = fsig(zf), ii = fsig(zi), o = fsig(zo);
      const float c = f * cbuf[i] + ii * tanhf(zg);
      cbuf[i] = c;
      const float h = o * tanhf(c);
      const int col = g * 16 + n;
      out[((size_t)t * BB + b) * HH + col] = h;            // h_seq
      hout[(size_t)b * HH + col] = (__bf16)h;              // next-step h (bf16)
      if (t == TT - 1) {
        out[(size_t)TT * BB * HH + (size_t)b * HH + col] = h;              // h final
        out[(size_t)TT * BB * HH + (size_t)BB * HH + (size_t)b * HH + col] = c; // c final
      }
    }
    __syncthreads();

    // ---- monotonic grid barrier (release h writes, acquire others')
    if (tid == 0) {
      __hip_atomic_fetch_add(cnt, 1u, __ATOMIC_RELEASE, __HIP_MEMORY_SCOPE_AGENT);
      const unsigned target = (unsigned)(t + 1) * (unsigned)NWG2;
      while (__hip_atomic_load(cnt, __ATOMIC_ACQUIRE, __HIP_MEMORY_SCOPE_AGENT) < target)
        __builtin_amdgcn_s_sleep(8);
    }
    __syncthreads();
  }
}

// ---------------------------------------------------------------------------
extern "C" void kernel_launch(void* const* d_in, const int* in_sizes, int n_in,
                              void* d_out, int out_size, void* d_ws, size_t ws_size,
                              hipStream_t stream) {
  const float* x   = (const float*)d_in[0];
  const float* Wxf = (const float*)d_in[1];  const float* bxf = (const float*)d_in[2];
  const float* Whf = (const float*)d_in[3];  const float* bhf = (const float*)d_in[4];
  const float* Wxi = (const float*)d_in[5];  const float* bxi = (const float*)d_in[6];
  const float* Whi = (const float*)d_in[7];  const float* bhi = (const float*)d_in[8];
  const float* Wxo = (const float*)d_in[9];  const float* bxo = (const float*)d_in[10];
  const float* Who = (const float*)d_in[11]; const float* bho = (const float*)d_in[12];
  const float* Wxg = (const float*)d_in[13]; const float* bxg = (const float*)d_in[14];
  const float* Whg = (const float*)d_in[15]; const float* bhg = (const float*)d_in[16];
  float* out = (float*)d_out;

  char* ws = (char*)d_ws;
  __bf16*   WxB   = (__bf16*)(ws + WS_WX);
  __bf16*   WhB   = (__bf16*)(ws + WS_WH);
  float*    bsum  = (float*)(ws + WS_BS);
  __bf16*   hping = (__bf16*)(ws + WS_H);
  unsigned* cnt   = (unsigned*)(ws + WS_CNT);
  __bf16*   xb    = (__bf16*)(ws + WS_XB);
  float*    xp    = (float*)(ws + WS_XP);

  lstm_prep<<<dim3((NG * NIN + 255) / 256), dim3(256), 0, stream>>>(
      Wxf, Whf, Wxi, Whi, Wxo, Who, Wxg, Whg,
      bxf, bhf, bxi, bhi, bxo, bho, bxg, bhg,
      WxB, WhB, bsum, hping, cnt);

  lstm_conv_x<<<dim3((size_t)MTOT * NIN / (256 * 8)), dim3(256), 0, stream>>>(x, xb);

  lstm_xp_gemm<<<dim3(MTOT / 128, NG / 128), dim3(256), 0, stream>>>(
      xb, WxB, bsum, xp);

  const size_t shmem = 64 * 512 * 2 + 4 * BB * 16 * 4 + BB * 16 * 4; // 84 KB
  lstm_recurrence<<<dim3(NWG2), dim3(256), shmem, stream>>>(
      xp, WhB, hping, out, cnt);
}
